// RobertaSelfAttention_matchKV_49340584296638
// MI455X (gfx1250) — matrix operand order repaired and checked by reading, not compiled, not run on hardware
//
#include <hip/hip_runtime.h>
#include <hip/hip_bf16.h>

// ---------------------------------------------------------------------------
// Problem dims (from reference)
// ---------------------------------------------------------------------------
#define BS   16
#define LEN  512
#define HID  768
#define NH   12
#define HD   64
#define NREG 8
#define R2   (2 * NREG)          // 16
#define MTOT (BS * LEN)          // 8192
#define NTOT (NH * HD)           // 768
#define KTOT HID                 // 768

// GEMM tiling
#define TM 128
#define TN 128
#define TK 32
#define LDSS 40                  // padded row stride in bf16 elems (80B = 20 banks)
#define TILE_ELEMS (TM * LDSS)   // 5120 ushorts = 10240 B per tile buffer
#define NKT (KTOT / TK)          // 24 K-steps

typedef __attribute__((ext_vector_type(16))) __bf16 v16bf;
typedef __attribute__((ext_vector_type(8)))  float  v8f;

union FragU {
    v16bf v;
    uint4 q[2];
};

__device__ __forceinline__ unsigned short f32_to_bf16_rne(float f) {
    unsigned int x = __float_as_uint(f);
    x += 0x7FFFu + ((x >> 16) & 1u);   // round-to-nearest-even
    return (unsigned short)(x >> 16);
}

// CDNA5 async global->LDS copy (ASYNCcnt-tracked, no VGPR round trip).
__device__ __forceinline__ void async_b128_to_lds(unsigned lds_addr,
                                                  const void* gaddr) {
    asm volatile("global_load_async_to_lds_b128 %0, %1, off"
                 :: "v"(lds_addr), "v"(gaddr)
                 : "memory");
}

__device__ __forceinline__ void wait_asynccnt0() {
    asm volatile("s_wait_asynccnt 0" ::: "memory");
}

// ---------------------------------------------------------------------------
// Kernel 1: f32 -> bf16 conversion (one-time, amortized across both GEMMs)
// ---------------------------------------------------------------------------
__global__ void cvt_bf16_kernel(const float* __restrict__ in,
                                unsigned short* __restrict__ out, int n) {
    int i = blockIdx.x * blockDim.x + threadIdx.x;
    if (i < n) out[i] = f32_to_bf16_rne(in[i]);
}

// ---------------------------------------------------------------------------
// Kernel 2: fused dual GEMM  C = relu(A * W^T + b)   (z=0: K1, z=1: V1)
// A: (MTOT x KTOT) bf16 row-major, W: (NTOT x KTOT) bf16 row-major.
// 128x128 tile per workgroup, 256 threads = 8 waves (wave32),
// each wave: 64x32 sub-tile = 4x2 WMMA 16x16 tiles, K stepped by 32.
// Double-buffered LDS fed by GLOBAL_LOAD_ASYNC_TO_LDS_B128; one barrier
// per K-step; async stage of tile k+1 overlaps the WMMAs of tile k.
// ---------------------------------------------------------------------------
__global__ __launch_bounds__(256)
void gemm_bf16_relu_kernel(const unsigned short* __restrict__ A,
                           const unsigned short* __restrict__ Kw,
                           const unsigned short* __restrict__ Vw,
                           const float* __restrict__ Kb,
                           const float* __restrict__ Vb,
                           float* __restrict__ K1,
                           float* __restrict__ V1) {
    __shared__ __align__(16) unsigned short sA[2 * TILE_ELEMS];
    __shared__ __align__(16) unsigned short sB[2 * TILE_ELEMS];

    const int z = blockIdx.z;
    const unsigned short* __restrict__ W = z ? Vw : Kw;
    const float* __restrict__ bias       = z ? Vb : Kb;
    float* __restrict__ C                = z ? V1 : K1;

    const int m0  = blockIdx.x * TM;
    const int n0  = blockIdx.y * TN;
    const int tid = threadIdx.x;
    const int wave = tid >> 5;
    const int lane = tid & 31;
    const int wy = wave & 1;    // 2 groups of 64 rows
    const int wx = wave >> 1;   // 4 groups of 32 cols
    const int mlane = lane & 15;
    const int khalf = lane >> 4;

    // LDS byte addresses (generic pointer low 32 bits = LDS offset)
    const unsigned ldsA = (unsigned)(size_t)(void*)sA;
    const unsigned ldsB = (unsigned)(size_t)(void*)sB;

    v8f acc[4][2] = {};

    // async-stage one 128x32 A tile + 128x32 W tile into buffer `buf`
    auto stage = [&](int buf, int kt) {
        const unsigned aBase = ldsA + (unsigned)(buf * TILE_ELEMS * 2);
        const unsigned bBase = ldsB + (unsigned)(buf * TILE_ELEMS * 2);
        #pragma unroll
        for (int i = tid; i < (TM * TK) / 8; i += 256) {   // 512 b128 chunks each
            const int row  = i >> 2;
            const int c8   = (i & 3) * 8;
            const unsigned loff = (unsigned)((row * LDSS + c8) * 2);
            async_b128_to_lds(aBase + loff,
                              A + (size_t)(m0 + row) * KTOT + kt + c8);
            async_b128_to_lds(bBase + loff,
                              W + (size_t)(n0 + row) * KTOT + kt + c8);
        }
    };

    stage(0, 0);

    for (int k = 0; k < NKT; ++k) {
        wait_asynccnt0();      // this wave's copies for tile k are in LDS
        __syncthreads();       // everyone's copies are in LDS

        if (k + 1 < NKT) stage((k + 1) & 1, (k + 1) * TK);  // overlaps WMMAs

        const unsigned short* sAb = sA + (k & 1) * TILE_ELEMS;
        const unsigned short* sBb = sB + (k & 1) * TILE_ELEMS;

        // A fragments: lane = row M, chunks at K = khalf*8 and 16 + khalf*8
        v16bf afr[4];
        #pragma unroll
        for (int i = 0; i < 4; ++i) {
            const unsigned short* p =
                sAb + (wy * 64 + i * 16 + mlane) * LDSS + khalf * 8;
            FragU u;
            u.q[0] = *(const uint4*)(p);
            u.q[1] = *(const uint4*)(p + 16);
            afr[i] = u.v;
        }
        // B fragments: lane = column N, chunks at K = khalf*8 and 16 + khalf*8
        v16bf bfr[2];
        #pragma unroll
        for (int j = 0; j < 2; ++j) {
            const unsigned short* p =
                sBb + (wx * 32 + j * 16 + mlane) * LDSS + khalf * 8;
            FragU u;
            u.q[0] = *(const uint4*)(p);
            u.q[1] = *(const uint4*)(p + 16);
            bfr[j] = u.v;
        }

        #pragma unroll
        for (int i = 0; i < 4; ++i)
            #pragma unroll
            for (int j = 0; j < 2; ++j)
                acc[i][j] = __builtin_amdgcn_wmma_f32_16x16x32_bf16(
                    false, afr[i], false, bfr[j], (short)0, acc[i][j],
                    false, false);
    }

    // epilogue: bias + relu; C/D layout: lane = col N, vgpr r = row r (+8 hi half)
    #pragma unroll
    for (int i = 0; i < 4; ++i) {
        const int mbase = m0 + wy * 64 + i * 16 + khalf * 8;
        #pragma unroll
        for (int j = 0; j < 2; ++j) {
            const int n = n0 + wx * 32 + j * 16 + mlane;
            const float b = bias[n];
            #pragma unroll
            for (int r = 0; r < 8; ++r) {
                C[(size_t)(mbase + r) * NTOT + n] = fmaxf(acc[i][j][r] + b, 0.0f);
            }
        }
    }
}

// ---------------------------------------------------------------------------
// Kernel 3: per (b,h): scores = K1 . head, softmax-over-L (+ 1/exp(max)
// residual), event threshold, cummax scans (fwd/bwd), 16-deep shift-register
// scan -> idx[(b*NH+h)*LEN*R2 + l*R2 + r]
// ---------------------------------------------------------------------------
__global__ __launch_bounds__(LEN)
void scores_scan_kernel(const float* __restrict__ K1,
                        const float* __restrict__ head,
                        int* __restrict__ idx) {
    const int bh = blockIdx.x;
    const int b  = bh / NH;
    const int h  = bh % NH;
    const int l  = threadIdx.x;

    __shared__ float red[LEN];
    __shared__ float shead[HD];
    __shared__ unsigned char ev[LEN];
    __shared__ int col0[LEN];
    __shared__ int col1[LEN];

    if (l < HD) shead[l] = head[h * HD + l];
    __syncthreads();

    // score: dot(K1[b,l,h,:], head[h,:])
    const float* kp = K1 + (size_t)(b * LEN + l) * NTOT + h * HD;
    float s = 0.0f;
    #pragma unroll 8
    for (int d = 0; d < HD; ++d) s += kp[d] * shead[d];

    // block max over L
    red[l] = s;
    __syncthreads();
    for (int off = LEN / 2; off > 0; off >>= 1) {
        if (l < off) red[l] = fmaxf(red[l], red[l + off]);
        __syncthreads();
    }
    const float m = red[0];
    __syncthreads();

    // block sum of exp
    const float e = __expf(s - m);
    red[l] = e;
    __syncthreads();
    for (int off = LEN / 2; off > 0; off >>= 1) {
        if (l < off) red[l] += red[l + off];
        __syncthreads();
    }
    const float sum = red[0];

    const float prob = e / sum + __expf(-m);
    ev[l] = (prob > (1.5f / (float)LEN)) ? 1 : 0;
    __syncthreads();

    // two serial scans on separate waves (cheap: 192 blocks x 512 steps)
    if (l == 0) {
        int last = -1;
        for (int i = 0; i < LEN; ++i) {
            if (ev[i]) last = i;
            col0[i] = (last >= 0) ? last : i;
        }
    } else if (l == 32) {
        // rolled = roll(ev,-1); fe[j] = ev[(LEN-j) % LEN];
        // col1[LEN-1-j] = LEN - cummax_idx(fe)[j]; col1[0] = 0
        int last = -1;
        for (int j = 0; j < LEN; ++j) {
            const int src = (LEN - j) & (LEN - 1);
            if (ev[src]) last = j;
            const int cm = (last >= 0) ? last : j;
            col1[LEN - 1 - j] = LEN - cm;
        }
        col1[0] = 0;
    }
    __syncthreads();

    // shift-register scan: reg shifts in [f_{l-1}, b_{l-1}] on event at l
    if (l == 0) {
        int reg[R2];
        #pragma unroll
        for (int r = 0; r < R2; ++r) reg[r] = 0;
        int* out = idx + (size_t)bh * LEN * R2;
        #pragma unroll
        for (int r = 0; r < R2; ++r) out[r] = 0;     // l = 0 row is zeros
        int pf = col0[0], pb = col1[0];
        for (int i = 1; i < LEN; ++i) {
            if (ev[i]) {
                #pragma unroll
                for (int r = R2 - 1; r >= 2; --r) reg[r] = reg[r - 2];
                reg[0] = pf;
                reg[1] = pb;
            }
            int* o = out + i * R2;
            #pragma unroll
            for (int r = 0; r < R2; ++r) {
                int v = reg[r];
                o[r] = (v < 0) ? 0 : ((v > LEN - 1) ? (LEN - 1) : v);
            }
            pf = col0[i];
            pb = col1[i];
        }
    }
}

// ---------------------------------------------------------------------------
// Kernel 4: out[b,l,h,d] = sum_r w[h,r] * V1[b, idx[b,h,l,r], h, d]
// 256 threads = 4 heads x 64 lanes-of-d per block.
// ---------------------------------------------------------------------------
__global__ __launch_bounds__(256)
void gather_mix_kernel(const float* __restrict__ V1,
                       const int* __restrict__ idx,
                       const float* __restrict__ bw,
                       float* __restrict__ out) {
    const int t  = threadIdx.x;
    const int d  = t & (HD - 1);
    const int hq = t >> 6;                    // 0..3
    const int g  = blockIdx.x;                // 0 .. BS*LEN*3 - 1
    const int h  = (g % 3) * 4 + hq;
    const int bl = g / 3;                     // b*LEN + l
    const int b  = bl >> 9;
    const int l  = bl & (LEN - 1);

    const int* ip = idx + ((size_t)(b * NH + h) * LEN + l) * R2;
    float acc = 0.0f;
    #pragma unroll
    for (int r = 0; r < R2; ++r) {
        const int i = ip[r];
        acc += bw[h * R2 + r] * V1[(size_t)(b * LEN + i) * NTOT + h * HD + d];
    }
    out[(size_t)bl * NTOT + h * HD + d] = acc;
}

// ---------------------------------------------------------------------------
// Launch
// ---------------------------------------------------------------------------
extern "C" void kernel_launch(void* const* d_in, const int* in_sizes, int n_in,
                              void* d_out, int out_size, void* d_ws, size_t ws_size,
                              hipStream_t stream) {
    const float* hs   = (const float*)d_in[0];  // (BS, LEN, HID)
    const float* K1w  = (const float*)d_in[1];  // (NTOT, HID)
    const float* K1b  = (const float*)d_in[2];  // (NTOT)
    const float* V1w  = (const float*)d_in[3];  // (NTOT, HID)
    const float* V1b  = (const float*)d_in[4];  // (NTOT)
    const float* head = (const float*)d_in[5];  // (NH, HD)
    const float* bw   = (const float*)d_in[6];  // (NH * R2)

    char* ws = (char*)d_ws;
    // workspace layout (256B aligned offsets)
    unsigned short* Abf  = (unsigned short*)(ws + 0);            // 12,582,912 B
    unsigned short* Kwbf = (unsigned short*)(ws + 12582912);     //  1,179,648 B
    unsigned short* Vwbf = (unsigned short*)(ws + 13762560);     //  1,179,648 B
    float*          K1   = (float*)(ws + 14942208);              // 25,165,824 B
    float*          V1   = (float*)(ws + 40108032);              // 25,165,824 B
    int*            idx  = (int*)(ws + 65273856);                //  6,291,456 B

    // 1) one-time bf16 conversion
    {
        int nA = MTOT * KTOT;
        cvt_bf16_kernel<<<(nA + 255) / 256, 256, 0, stream>>>(hs, Abf, nA);
        int nW = NTOT * KTOT;
        cvt_bf16_kernel<<<(nW + 255) / 256, 256, 0, stream>>>(K1w, Kwbf, nW);
        cvt_bf16_kernel<<<(nW + 255) / 256, 256, 0, stream>>>(V1w, Vwbf, nW);
    }

    // 2) dual WMMA GEMM (z=0 -> K1, z=1 -> V1), bias + relu fused
    {
        dim3 grid(MTOT / TM, NTOT / TN, 2);   // 64 x 6 x 2
        gemm_bf16_relu_kernel<<<grid, 256, 0, stream>>>(Abf, Kwbf, Vwbf,
                                                        K1b, V1b, K1, V1);
    }

    // 3) scores + softmax + event scans + shift-register indices
    scores_scan_kernel<<<BS * NH, LEN, 0, stream>>>(K1, head, idx);

    // 4) gather + bidirection-weight mix -> output (BS, LEN, NH, HD) f32
    gather_mix_kernel<<<(BS * LEN * NH) / 4, 256, 0, stream>>>(V1, idx, bw,
                                                               (float*)d_out);
}
